// RNN_41403484733766
// MI455X (gfx1250) — matrix operand order repaired and checked by reading, not compiled, not run on hardware
//
#include <hip/hip_runtime.h>
#include <math.h>

// ---------------------------------------------------------------------------
// 2-layer GELU RNN for MI455X (gfx1250, wave32), two phases:
//
//  Phase 1 (parallel WMMA GEMM): xin0 = seq @ W_in0^T, written into d_out's
//    out1 region (xin0[t] occupies exactly the slot out1[t] later overwrites).
//  Phase 2 (persistent, 16 blocks): fused sequential recurrence for both
//    layers. Layer0: h0 <- gelu(xin0[t] + h0 @ Wh0^T) (C-init from xin0).
//    Layer1: h1 <- gelu([h0' | h1] @ [Win1;Wh1]^T) (concat-GEMM).
//    Weights live in VGPRs; LDS holds only ping-pong h panels; 2 barriers/step.
// ---------------------------------------------------------------------------

typedef __attribute__((ext_vector_type(16))) _Float16 v16h;
typedef __attribute__((ext_vector_type(8)))  _Float16 v8h;
typedef __attribute__((ext_vector_type(8)))  float    v8f;

#define T_SEQ 1024
#define NBATCH 256
#define IN0   64
#define HID   128
#define KH    4      // HID/32 K-chunks (recurrent part, both layers)
#define KC1   8      // (HID+HID)/32 K-chunks, layer 1 concat
// A0 row: [ h0_even:0..127 | h0_odd:128..255 ]            stride 264 f16
// A1 row: [ x:0..127 | h1_even:128..255 | h1_odd:256..383 ] stride 392 f16
#define A0S   264
#define A1S   392

// Branchless exact-GELU: 0.5*x*(1+erf(x/sqrt2)), erf via A&S 7.1.26
// (|err| <= 1.5e-7 << f16 WMMA rounding). No EXEC divergence, 2 TRANS ops.
__device__ __forceinline__ float gelu_exact(float x) {
    const float z  = x * 0.70710678118654752f;
    const float az = __builtin_fabsf(z);
    const float t  = __builtin_amdgcn_rcpf(__builtin_fmaf(0.3275911f, az, 1.0f));
    float p = __builtin_fmaf(1.061405429f, t, -1.453152027f);
    p = __builtin_fmaf(p, t,  1.421413741f);
    p = __builtin_fmaf(p, t, -0.284496736f);
    p = __builtin_fmaf(p, t,  0.254829592f);
    p = p * t;
    const float e = __builtin_amdgcn_exp2f(az * az * -1.4426950408889634f); // e^{-z^2}
    const float erf_abs = __builtin_fmaf(-p, e, 1.0f);
    const float erf_z   = __builtin_copysignf(erf_abs, z);
    return 0.5f * x * (1.0f + erf_z);
}

__device__ __forceinline__ v16h load_frag(const _Float16* p) {
    v8h alo = *(const v8h*)p;
    v8h ahi = *(const v8h*)(p + 8);
    return __builtin_shufflevector(alo, ahi,
                                   0, 1, 2, 3, 4, 5, 6, 7,
                                   8, 9, 10, 11, 12, 13, 14, 15);
}

// ------------------------- Phase 1: xin0 GEMM ------------------------------
// xout[m][n] = sum_k seq[m][k] * Win0[n][k], m over T*B rows. 8 waves/block,
// one 16-row M-strip per wave, full N=128, K=64 (2 chunks).
__global__ __launch_bounds__(256)
void xproj0_wmma(const float* __restrict__ seq, const float* __restrict__ Win0,
                 float* __restrict__ xout)
{
    const int tid = threadIdx.x;
    const int lane = tid & 31;
    const int wave = tid >> 5;
    const int nl = lane & 15;
    const bool hiHalf = (lane >= 16);
    const int halfSel = hiHalf ? 16 : 0;
    const int mOff = hiHalf ? 8 : 0;
    const int mBase = blockIdx.x * 128 + wave * 16;

    // B fragments: 8 N-tiles x 2 K-chunks (f32 -> f16), weights are L2-hot.
    v16h Bf[8][2];
    for (int nt = 0; nt < 8; ++nt)
        for (int kc = 0; kc < 2; ++kc) {
            const int k0 = kc * 32 + (hiHalf ? 16 : 0);
            v16h f;
            for (int j = 0; j < 16; ++j)
                f[j] = (_Float16)Win0[(nt * 16 + nl) * IN0 + k0 + j];
            Bf[nt][kc] = f;
        }

    // A fragments: row mBase + (lane&15), two 32-wide K chunks.
    v16h a[2];
    const float* arow = seq + (size_t)(mBase + nl) * IN0;
#pragma unroll
    for (int kc = 0; kc < 2; ++kc) {
        const int kb = kc * 32 + halfSel;
        v16h f;
#pragma unroll
        for (int j = 0; j < 8; ++j) f[j]     = (_Float16)arow[kb + j];
#pragma unroll
        for (int j = 0; j < 8; ++j) f[8 + j] = (_Float16)arow[kb + 8 + j];
        a[kc] = f;
    }

    float* orow = xout + (size_t)(mBase + mOff) * HID + nl;
#pragma unroll
    for (int nt = 0; nt < 8; ++nt) {
        v8f c = {};
        c = __builtin_amdgcn_wmma_f32_16x16x32_f16(false, a[0], false, Bf[nt][0],
                                                   (short)0, c, false, false);
        c = __builtin_amdgcn_wmma_f32_16x16x32_f16(false, a[1], false, Bf[nt][1],
                                                   (short)0, c, false, false);
#pragma unroll
        for (int r = 0; r < 8; ++r)
            orow[r * HID + nt * 16] = c[r];
    }
}

// ------------------- Phase 2: fused sequential recurrence ------------------
__global__ __launch_bounds__(256, 1)
void rnn2_fused_wmma(const float* __restrict__ Wh0,
                     const float* __restrict__ Win1, const float* __restrict__ Wh1,
                     const float* __restrict__ h00,  const float* __restrict__ h01,
                     float* __restrict__ out)
{
    __shared__ __align__(16) _Float16 A0[16 * A0S];
    __shared__ __align__(16) _Float16 A1[16 * A1S];

    const int tid    = threadIdx.x;
    const int lane   = tid & 31;
    const int wave   = tid >> 5;             // 0..7 -> N-tile
    const int bBase  = blockIdx.x * 16;      // batch tile origin
    const int nl     = lane & 15;
    const bool hiHalf = (lane >= 16);
    const int n      = wave * 16 + nl;       // global output column 0..127
    const int halfSel = hiHalf ? 16 : 0;
    const int am     = nl;                   // A-fragment row (M = lane % 16)
    const int mOff   = hiHalf ? 8 : 0;       // C/D row offset per half-wave

    // -------- weight B-fragments into registers, once --------
    v16h B0[KH];                 // Wh0 only (x-projection precomputed)
    v16h B1[KC1];                // [Win1 ; Wh1]
    for (int kc = 0; kc < KH; ++kc) {
        const int k0 = kc * 32 + (hiHalf ? 16 : 0);
        v16h f;
        for (int j = 0; j < 16; ++j)
            f[j] = (_Float16)Wh0[n * HID + k0 + j];
        B0[kc] = f;
    }
    for (int kc = 0; kc < KC1; ++kc) {
        const int k0 = kc * 32 + (hiHalf ? 16 : 0);
        v16h f;
        for (int j = 0; j < 16; ++j) {
            const int k = k0 + j;
            const float w = (k < HID) ? Win1[n * HID + k] : Wh1[n * HID + (k - HID)];
            f[j] = (_Float16)w;
        }
        B1[kc] = f;
    }

    // -------- init: h params broadcast into the "even" (t=0) h regions --------
    for (int i = tid; i < 16 * HID; i += 256) {
        const int m = i >> 7, k = i & (HID - 1);
        A0[m * A0S + k]       = (_Float16)h00[k];
        A1[m * A1S + HID + k] = (_Float16)h01[k];
    }

    // -------- strength-reduced global pointers (one u64 add per step) --------
    float* o1 = out + (size_t)(bBase + mOff) * HID + n;   // xin0[t] in, out1[t] out
    float* hn0 = out + (size_t)T_SEQ * NBATCH * HID + (size_t)(bBase + mOff) * HID + n;
    float* hn1 = hn0 + (size_t)NBATCH * HID;
    const _Float16* ra0 = &A0[am * A0S + halfSel];
    const _Float16* ra1 = &A1[am * A1S + halfSel];
    _Float16* w0 = &A0[mOff * A0S + n];
    _Float16* w1 = &A1[mOff * A1S + n];

    // xin0[t=0]: no barrier dependency, load ahead.
    v8f xb;
#pragma unroll
    for (int r = 0; r < 8; ++r) xb[r] = o1[r * HID];

    for (int t = 0; t < T_SEQ; ++t) {
        const int hoff0 = (t & 1) * HID;            // current h phase offset
        const int hdst0 = HID - hoff0;              // h0' dest (opposite phase)
        const int hdst1 = 2 * HID - hoff0;          // h1' dest
        __syncthreads();   // B_a: new h panels visible to all readers

        // ---- hoist recurrent A-fragment loads (distinct regs -> pipelined) ----
        v16h a0[KH];
#pragma unroll
        for (int kc = 0; kc < KH; ++kc)
            a0[kc] = load_frag(ra0 + kc * 32 + hoff0);
        v16h a1h[KH];              // layer-1 h fragments, also valid after B_a
#pragma unroll
        for (int kc = 0; kc < KH; ++kc)
            a1h[kc] = load_frag(ra1 + (kc + 4) * 32 + hoff0);

        // ---- layer 0: C0 = xin0[t] + h0 @ Wh0^T  (C-init from xb) ----
        v8f c0 = xb;
#pragma unroll
        for (int kc = 0; kc < KH; ++kc)
            c0 = __builtin_amdgcn_wmma_f32_16x16x32_f16(false, a0[kc], false, B0[kc],
                                                        (short)0, c0, false, false);

        // ---- gelu -> h0' (opposite phase) and layer-1 x region ----
#pragma unroll
        for (int r = 0; r < 8; ++r) {
            const float g = gelu_exact(c0[r]);
            const _Float16 hv = (_Float16)g;
            w0[r * A0S + hdst0] = hv;
            w1[r * A1S]         = hv;
            if (t == T_SEQ - 1)              // hn[0]
                hn0[r * HID] = g;
        }
        __syncthreads();   // B_c: out0 tile visible before layer-1 reads

        // ---- prefetch next step's xin0 (hides behind whole L1 phase) ----
        if (t + 1 < T_SEQ) {
#pragma unroll
            for (int r = 0; r < 8; ++r)
                xb[r] = o1[(size_t)NBATCH * HID + r * HID];
        }

        // ---- layer 1: x-fragment loads first, consume h-chunks while in flight --
        v16h a1x[4];
#pragma unroll
        for (int kc = 0; kc < 4; ++kc)
            a1x[kc] = load_frag(ra1 + kc * 32);

        v8f c1 = {};
#pragma unroll
        for (int kc = 0; kc < KH; ++kc)        // h chunks: registers, no wait
            c1 = __builtin_amdgcn_wmma_f32_16x16x32_f16(false, a1h[kc], false, B1[kc + 4],
                                                        (short)0, c1, false, false);
#pragma unroll
        for (int kc = 0; kc < 4; ++kc)         // x chunks: loads had time to land
            c1 = __builtin_amdgcn_wmma_f32_16x16x32_f16(false, a1x[kc], false, B1[kc],
                                                        (short)0, c1, false, false);

        // ---- gelu -> out1 (overwrites consumed xin0[t]) and h1' ----
#pragma unroll
        for (int r = 0; r < 8; ++r) {
            const float g = gelu_exact(c1[r]);
            w1[r * A1S + hdst1] = (_Float16)g;
            o1[r * HID] = g;
            if (t == T_SEQ - 1)              // hn[1]
                hn1[r * HID] = g;
        }
        o1 += (size_t)NBATCH * HID;
    }
}

extern "C" void kernel_launch(void* const* d_in, const int* in_sizes, int n_in,
                              void* d_out, int out_size, void* d_ws, size_t ws_size,
                              hipStream_t stream) {
    (void)in_sizes; (void)n_in; (void)d_ws; (void)ws_size; (void)out_size;
    const float* seq  = (const float*)d_in[0];
    const float* Win0 = (const float*)d_in[1];
    const float* Wh0  = (const float*)d_in[2];
    const float* Win1 = (const float*)d_in[3];
    const float* Wh1  = (const float*)d_in[4];
    const float* h00  = (const float*)d_in[5];
    const float* h01  = (const float*)d_in[6];
    float* out = (float*)d_out;

    // Phase 1: parallel input projection into the out1 region of d_out.
    xproj0_wmma<<<dim3((T_SEQ * NBATCH) / 128), dim3(256), 0, stream>>>(seq, Win0, out);
    // Phase 2: persistent fused recurrence (16 batch tiles).
    rnn2_fused_wmma<<<dim3(NBATCH / 16), dim3(256), 0, stream>>>(Wh0, Win1, Wh1,
                                                                 h00, h01, out);
}
